// StateSpaceDualityBlock_30107720745665
// MI455X (gfx1250) — compile-verified, matrix operand order checked by reading, not run on hardware
//
#include <hip/hip_runtime.h>
#include <cstdint>
#include <cstddef>

// ---- Problem dimensions (from reference) ----
#define D_MODEL   256
#define HEADDIM   32
#define D_STATE   64
#define D_CONV    4
#define D_INNER   512          // EXPAND * D_MODEL
#define NHEADS    16           // D_INNER / HEADDIM
#define CONV_DIM  640          // D_INNER + 2*D_STATE
#define D_IN_PROJ 1168         // 2*D_INNER + 2*D_STATE + NHEADS
#define BATCH     8
#define SEQ       2048
#define NROWS     (BATCH * SEQ)   // 16384
#define EPS       1e-5f

typedef float    v2f  __attribute__((ext_vector_type(2)));
typedef float    v8f  __attribute__((ext_vector_type(8)));
typedef _Float16 v16h __attribute__((ext_vector_type(16)));

// fp32 WMMA (V_WMMA_F32_16X16X4_F32) confirmed to lower on gfx1250.
#if defined(__AMDGCN__) && __has_builtin(__builtin_amdgcn_wmma_f32_16x16x4_f32)
#define USE_F32_WMMA 1
#endif

// =====================================================================
// GEMM: C[M,N] = A[M,K] * B[N,K]^T  (row-major, arbitrary leading dims)
// One wave32 per 64(M) x 16(N) output block = 4 WMMA tiles stacked in M.
// Each B fragment is reused by 4 WMMAs -> 1.25 loads per v_wmma.
// Requires: M % 64 == 0, N % 16 == 0, K % 32 == 0.
// A-frag layout (f32 16x4): lanes 0-15 hold K=k..k+1, lanes 16-31 K=k+2..k+3.
// C/D layout: VGPR i holds row (i + 8*(lane>=16)), col = lane&15.
// =====================================================================
__global__ void gemm_abt_wmma(const float* __restrict__ A,
                              const float* __restrict__ B,
                              float* __restrict__ C,
                              int M, int N, int K,
                              int lda, int ldb, int ldc) {
    const int ntn   = N >> 4;
    const int wave  = blockIdx.x * (blockDim.x >> 5) + (threadIdx.x >> 5);
    const int total = (M >> 6) * ntn;          // 64-row blocks
    if (wave >= total) return;
    const int bm   = wave / ntn;               // 64-row block index
    const int tn   = wave % ntn;
    const int lane = threadIdx.x & 31;
    const int half = lane >> 4;                // 0: lanes 0-15, 1: lanes 16-31
    const int lo   = lane & 15;

    const float* Ap0 = A + (size_t)(bm * 64 + lo) * lda;        // tile 0 row
    const float* Ap1 = Ap0 + (size_t)16 * lda;
    const float* Ap2 = Ap0 + (size_t)32 * lda;
    const float* Ap3 = Ap0 + (size_t)48 * lda;
    const float* Bp  = B + (size_t)(tn * 16 + lo) * ldb;

    v8f acc0 = {}, acc1 = {}, acc2 = {}, acc3 = {};

#ifdef USE_F32_WMMA
    #pragma unroll 2
    for (int k = 0; k < K; k += 4) {
        const int ko = k + half * 2;
        // Prefetch next K-panel (lowers to global_prefetch_b8; speculative,
        // OOB prefetches are dropped by hardware).
        __builtin_prefetch(Ap0 + ko + 64, 0, 1);
        __builtin_prefetch(Ap2 + ko + 64, 0, 1);
        v2f b  = *(const v2f*)(Bp  + ko);
        v2f a0 = *(const v2f*)(Ap0 + ko);
        v2f a1 = *(const v2f*)(Ap1 + ko);
        v2f a2 = *(const v2f*)(Ap2 + ko);
        v2f a3 = *(const v2f*)(Ap3 + ko);
        acc0 = __builtin_amdgcn_wmma_f32_16x16x4_f32(false, a0, false, b, (short)0, acc0, false, false);
        acc1 = __builtin_amdgcn_wmma_f32_16x16x4_f32(false, a1, false, b, (short)0, acc1, false, false);
        acc2 = __builtin_amdgcn_wmma_f32_16x16x4_f32(false, a2, false, b, (short)0, acc2, false, false);
        acc3 = __builtin_amdgcn_wmma_f32_16x16x4_f32(false, a3, false, b, (short)0, acc3, false, false);
    }
#else
    for (int k = 0; k < K; k += 32) {
        const int kb = k + half * 8;           // 16-bit A 16x32 striping
        v16h b, a0, a1, a2, a3;
        #pragma unroll
        for (int j = 0; j < 8; ++j) {
            b[j]      = (_Float16)Bp[kb + j];
            b[8 + j]  = (_Float16)Bp[kb + 16 + j];
            a0[j]     = (_Float16)Ap0[kb + j];
            a0[8 + j] = (_Float16)Ap0[kb + 16 + j];
            a1[j]     = (_Float16)Ap1[kb + j];
            a1[8 + j] = (_Float16)Ap1[kb + 16 + j];
            a2[j]     = (_Float16)Ap2[kb + j];
            a2[8 + j] = (_Float16)Ap2[kb + 16 + j];
            a3[j]     = (_Float16)Ap3[kb + j];
            a3[8 + j] = (_Float16)Ap3[kb + 16 + j];
        }
        acc0 = __builtin_amdgcn_wmma_f32_16x16x32_f16(false, a0, false, b, (short)0, acc0, false, false);
        acc1 = __builtin_amdgcn_wmma_f32_16x16x32_f16(false, a1, false, b, (short)0, acc1, false, false);
        acc2 = __builtin_amdgcn_wmma_f32_16x16x32_f16(false, a2, false, b, (short)0, acc2, false, false);
        acc3 = __builtin_amdgcn_wmma_f32_16x16x32_f16(false, a3, false, b, (short)0, acc3, false, false);
    }
#endif

    float* Cp = C + (size_t)(bm * 64 + half * 8) * ldc + tn * 16 + lo;
    #pragma unroll
    for (int i = 0; i < 8; ++i) Cp[(size_t)(i)      * ldc] = acc0[i];
    #pragma unroll
    for (int i = 0; i < 8; ++i) Cp[(size_t)(i + 16) * ldc] = acc1[i];
    #pragma unroll
    for (int i = 0; i < 8; ++i) Cp[(size_t)(i + 32) * ldc] = acc2[i];
    #pragma unroll
    for (int i = 0; i < 8; ++i) Cp[(size_t)(i + 48) * ldc] = acc3[i];
}

// =====================================================================
// K2: depthwise causal conv (width 4) over xBC channels + bias + SiLU
// =====================================================================
__global__ void conv_silu_kernel(const float* __restrict__ zxbcdt,
                                 const float* __restrict__ conv_w,
                                 const float* __restrict__ conv_b,
                                 float* __restrict__ xbc) {
    int idx = blockIdx.x * blockDim.x + threadIdx.x;
    if (idx >= NROWS * CONV_DIM) return;
    int c   = idx % CONV_DIM;
    int row = idx / CONV_DIM;     // b*SEQ + t
    int t   = row % SEQ;
    const float* src = zxbcdt + (size_t)row * D_IN_PROJ + D_INNER + c;
    float acc = conv_b[c];
    #pragma unroll
    for (int j = 0; j < D_CONV; ++j) {
        int tt = t - (D_CONV - 1) + j;
        if (tt >= 0)
            acc = fmaf(src[(j - (D_CONV - 1)) * D_IN_PROJ],
                       conv_w[c * D_CONV + j], acc);
    }
    xbc[(size_t)row * CONV_DIM + c] = acc / (1.0f + __expf(-acc));
}

// =====================================================================
// K3: dt = softplus(dt_raw + bias); dA = exp(dt * (-exp(A_log)))
// =====================================================================
__global__ void dt_kernel(const float* __restrict__ zxbcdt,
                          const float* __restrict__ dt_bias,
                          const float* __restrict__ A_log,
                          float* __restrict__ dtbuf,
                          float* __restrict__ dAbuf) {
    int idx = blockIdx.x * blockDim.x + threadIdx.x;
    if (idx >= NROWS * NHEADS) return;
    int h   = idx % NHEADS;
    int row = idx / NHEADS;
    float v  = zxbcdt[(size_t)row * D_IN_PROJ + D_INNER + CONV_DIM + h] + dt_bias[h];
    float dt = (v > 20.0f) ? v : log1pf(__expf(v));
    float A  = -__expf(A_log[h]);
    dtbuf[idx] = dt;
    dAbuf[idx] = __expf(dt * A);
}

// =====================================================================
// K4: selective scan. One wave32 per (batch, head); lane = headdim index p.
// State h[p][n] (32x64) lives in 64 VGPRs per lane. Serial over T=2048.
// =====================================================================
__global__ void __launch_bounds__(32)
scan_kernel(const float* __restrict__ xbc,
            const float* __restrict__ dtbuf,
            const float* __restrict__ dAbuf,
            const float* __restrict__ Dvec,
            float* __restrict__ ybuf) {
    const int h = blockIdx.x;
    const int b = blockIdx.y;
    const int p = threadIdx.x;    // 0..31 == headdim
    float hs[D_STATE];
    #pragma unroll
    for (int n = 0; n < D_STATE; ++n) hs[n] = 0.0f;
    const float Dh = Dvec[h];

    for (int t = 0; t < SEQ; ++t) {
        const size_t row = (size_t)(b * SEQ + t);
        const float* xr  = xbc + row * CONV_DIM;
        const float  xv  = xr[h * HEADDIM + p];         // per-lane
        const float  dtv = dtbuf[row * NHEADS + h];     // uniform -> scalar
        const float  dAv = dAbuf[row * NHEADS + h];     // uniform -> scalar
        const float* Br  = xr + D_INNER;                // uniform rows
        const float* Cr  = xr + D_INNER + D_STATE;
        const float coef = dtv * xv;
        float y = 0.0f;
        #pragma unroll
        for (int n = 0; n < D_STATE; ++n) {
            hs[n] = fmaf(hs[n], dAv, coef * Br[n]);
            y     = fmaf(hs[n], Cr[n], y);
        }
        ybuf[row * D_INNER + h * HEADDIM + p] = fmaf(Dh, xv, y);
    }
}

// =====================================================================
// K5a: y = (y * silu(z)); RMSNorm over 512 * norm_w (in-place in ybuf)
// =====================================================================
__global__ void gate_rms_kernel(const float* __restrict__ zxbcdt,
                                const float* __restrict__ norm_w,
                                float* __restrict__ ybuf) {
    const int row = blockIdx.x;
    const int tid = threadIdx.x;                 // 256 threads, 2 elems each
    const float* zr = zxbcdt + (size_t)row * D_IN_PROJ;
    float* yr = ybuf + (size_t)row * D_INNER;

    float z0 = zr[tid], z1 = zr[tid + 256];
    float g0 = yr[tid]       * (z0 / (1.0f + __expf(-z0)));
    float g1 = yr[tid + 256] * (z1 / (1.0f + __expf(-z1)));

    __shared__ float red[256];
    red[tid] = g0 * g0 + g1 * g1;
    __syncthreads();
    for (int s = 128; s > 0; s >>= 1) {
        if (tid < s) red[tid] += red[tid + s];
        __syncthreads();
    }
    float scale = rsqrtf(red[0] / (float)D_INNER + EPS);
    yr[tid]       = g0 * scale * norm_w[tid];
    yr[tid + 256] = g1 * scale * norm_w[tid + 256];
}

// =====================================================================
// K6: residual add + LayerNorm over 256 -> final output
// =====================================================================
__global__ void resid_ln_kernel(const float* __restrict__ obuf,
                                const float* __restrict__ x,
                                const float* __restrict__ ln_w,
                                const float* __restrict__ ln_b,
                                float* __restrict__ out) {
    const int row = blockIdx.x;
    const int tid = threadIdx.x;   // 256 threads, 1 elem each
    float r = obuf[(size_t)row * D_MODEL + tid] + x[(size_t)row * D_MODEL + tid];

    __shared__ float red[256];
    red[tid] = r;
    __syncthreads();
    for (int s = 128; s > 0; s >>= 1) {
        if (tid < s) red[tid] += red[tid + s];
        __syncthreads();
    }
    float mu = red[0] / (float)D_MODEL;
    __syncthreads();
    float d = r - mu;
    red[tid] = d * d;
    __syncthreads();
    for (int s = 128; s > 0; s >>= 1) {
        if (tid < s) red[tid] += red[tid + s];
        __syncthreads();
    }
    float inv = rsqrtf(red[0] / (float)D_MODEL + EPS);
    out[(size_t)row * D_MODEL + tid] = d * inv * ln_w[tid] + ln_b[tid];
}

// =====================================================================
extern "C" void kernel_launch(void* const* d_in, const int* in_sizes, int n_in,
                              void* d_out, int out_size, void* d_ws, size_t ws_size,
                              hipStream_t stream) {
    const float* x          = (const float*)d_in[0];
    const float* in_proj_w  = (const float*)d_in[1];
    const float* conv_w     = (const float*)d_in[2];
    const float* conv_b     = (const float*)d_in[3];
    const float* dt_bias    = (const float*)d_in[4];
    const float* A_log      = (const float*)d_in[5];
    const float* Dv         = (const float*)d_in[6];
    const float* norm_w     = (const float*)d_in[7];
    const float* out_proj_w = (const float*)d_in[8];
    const float* ln_w       = (const float*)d_in[9];
    const float* ln_b       = (const float*)d_in[10];
    float* out = (float*)d_out;
    (void)in_sizes; (void)n_in; (void)out_size; (void)ws_size;

    // Workspace layout (~170.9 MB total)
    char*  ws  = (char*)d_ws;
    size_t off = 0;
    auto alloc = [&](size_t bytes) -> float* {
        float* p = (float*)(ws + off);
        off += (bytes + 255) & ~(size_t)255;
        return p;
    };
    float* zxbcdt = alloc((size_t)NROWS * D_IN_PROJ * sizeof(float)); // 76.5 MB
    float* xbc    = alloc((size_t)NROWS * CONV_DIM  * sizeof(float)); // 41.9 MB
    float* dtbuf  = alloc((size_t)NROWS * NHEADS    * sizeof(float)); //  1.0 MB
    float* dAbuf  = alloc((size_t)NROWS * NHEADS    * sizeof(float)); //  1.0 MB
    float* ybuf   = alloc((size_t)NROWS * D_INNER   * sizeof(float)); // 33.5 MB
    float* obuf   = alloc((size_t)NROWS * D_MODEL   * sizeof(float)); // 16.8 MB

    // K1: zxbcdt = x @ in_proj_w^T   (M=16384, N=1168, K=256)
    {
        int waves  = (NROWS / 64) * (D_IN_PROJ / 16);   // 256*73 = 18688
        int blocks = (waves + 7) / 8;                   // 8 waves / block
        gemm_abt_wmma<<<blocks, 256, 0, stream>>>(
            x, in_proj_w, zxbcdt, NROWS, D_IN_PROJ, D_MODEL,
            D_MODEL, D_MODEL, D_IN_PROJ);
    }
    // K2: depthwise conv + SiLU
    {
        int n = NROWS * CONV_DIM;
        conv_silu_kernel<<<(n + 255) / 256, 256, 0, stream>>>(
            zxbcdt, conv_w, conv_b, xbc);
    }
    // K3: dt / dA
    {
        int n = NROWS * NHEADS;
        dt_kernel<<<(n + 255) / 256, 256, 0, stream>>>(
            zxbcdt, dt_bias, A_log, dtbuf, dAbuf);
    }
    // K4: selective scan (128 independent streams, one wave32 each)
    scan_kernel<<<dim3(NHEADS, BATCH), 32, 0, stream>>>(
        xbc, dtbuf, dAbuf, Dv, ybuf);
    // K5a: gate + RMSNorm (in-place)
    gate_rms_kernel<<<NROWS, 256, 0, stream>>>(zxbcdt, norm_w, ybuf);
    // K5b: obuf = ybuf @ out_proj_w^T  (M=16384, N=256, K=512)
    {
        int waves  = (NROWS / 64) * (D_MODEL / 16);     // 256*16 = 4096
        int blocks = (waves + 7) / 8;
        gemm_abt_wmma<<<blocks, 256, 0, stream>>>(
            ybuf, out_proj_w, obuf, NROWS, D_MODEL, D_INNER,
            D_INNER, D_INNER, D_MODEL);
    }
    // K6: residual + LayerNorm -> out
    resid_ln_kernel<<<NROWS, 256, 0, stream>>>(obuf, x, ln_w, ln_b, out);
}
